// SelfAttention_41223096107222
// MI455X (gfx1250) — matrix operand order repaired and checked by reading, not compiled, not run on hardware
//
#include <hip/hip_runtime.h>
#include <hip/hip_bf16.h>
#include <stdint.h>

#define B_  4
#define C_  256
#define N_  4096
#define KD_ 32
#define D_  256
#define WAVES 8

typedef __attribute__((ext_vector_type(16))) __bf16 v16bf;
typedef __attribute__((ext_vector_type(8)))  float  v8f;

union U32B { uint4 q[2]; v16bf v; };
union FU { float f; unsigned u; };

// v_permlane16_b32 xor-butterfly step within each 16-lane group (pure VALU).
__device__ __forceinline__ float perml16(float v, unsigned s0, unsigned s1) {
    FU a; a.f = v;
    FU r; r.u = __builtin_amdgcn_permlane16(a.u, a.u, s0, s1, true, false);
    return r.f;
}
__device__ __forceinline__ float rowmax16(float v) {
    v = fmaxf(v, perml16(v, 0x67452301u, 0xEFCDAB89u));   // xor 1
    v = fmaxf(v, perml16(v, 0x54761032u, 0xDCFE98BAu));   // xor 2
    v = fmaxf(v, perml16(v, 0x32107654u, 0xBA98FEDCu));   // xor 4
    v = fmaxf(v, perml16(v, 0xFEDCBA98u, 0x76543210u));   // xor 8
    return v;
}

// ---------------------------------------------------------------------------
// Kernel 1: f = (Wf x + bf) * log2(e)  -> Qb bf16 [B,N,KD]   (exp2-domain Q)
//           g =  Wg x + bg             -> Kb bf16 [B,N,KD]
// ---------------------------------------------------------------------------
__global__ __launch_bounds__(256)
void fg_kernel(const float* __restrict__ x, const float* __restrict__ Wf,
               const float* __restrict__ bf, const float* __restrict__ Wg,
               const float* __restrict__ bg, __bf16* __restrict__ Qb,
               __bf16* __restrict__ Kb) {
    int idx = blockIdx.x * blockDim.x + threadIdx.x;   // 0 .. B*N-1
    int b = idx >> 12;           // / N_
    int n = idx & (N_ - 1);
    const float* xb = x + (size_t)b * C_ * N_ + n;
    float fa[KD_], ga[KD_];
#pragma unroll
    for (int k = 0; k < KD_; ++k) { fa[k] = 0.f; ga[k] = 0.f; }
    for (int c = 0; c < C_; ++c) {
        float xv = xb[(size_t)c * N_];
#pragma unroll
        for (int k = 0; k < KD_; ++k) {
            fa[k] = fmaf(Wf[k * C_ + c], xv, fa[k]);   // uniform -> s_load
            ga[k] = fmaf(Wg[k * C_ + c], xv, ga[k]);
        }
    }
    __bf16* qr = Qb + (size_t)idx * KD_;
    __bf16* kr = Kb + (size_t)idx * KD_;
#pragma unroll
    for (int k = 0; k < KD_; ++k) {
        qr[k] = (__bf16)((fa[k] + bf[k]) * 1.44269504088896f);
        kr[k] = (__bf16)(ga[k] + bg[k]);
    }
}

// ---------------------------------------------------------------------------
// Kernel 2: t [B,N,D] f32  ->  Vt [B,D,N] bf16  (LDS-tiled transpose)
// ---------------------------------------------------------------------------
__global__ __launch_bounds__(256)
void vt_kernel(const float* __restrict__ t, __bf16* __restrict__ Vt) {
    __shared__ float sh[32][33];
    int b  = blockIdx.z;
    int n0 = blockIdx.x * 32;
    int d0 = blockIdx.y * 32;
    int tx = threadIdx.x, ty = threadIdx.y;            // 32 x 8
    const float* tb = t + (size_t)b * N_ * D_;
#pragma unroll
    for (int i = 0; i < 4; ++i)
        sh[ty + 8 * i][tx] = tb[(size_t)(n0 + ty + 8 * i) * D_ + d0 + tx];
    __syncthreads();
    __bf16* vb = Vt + (size_t)b * D_ * N_;
#pragma unroll
    for (int i = 0; i < 4; ++i)
        vb[(size_t)(d0 + ty + 8 * i) * N_ + n0 + tx] = (__bf16)sh[tx][ty + 8 * i];
}

// ---------------------------------------------------------------------------
// Kernel 3: flash attention. One wave = one 16-row query tile, full D=256 in
// registers. m-loop chunks of 64: 4 QK + 2 row-sum + 32 PV wmma per iter.
// Operands staged in arrays so loads clause up and waits stay partial.
// ---------------------------------------------------------------------------
__global__ __launch_bounds__(32 * WAVES)
void attn_kernel(const __bf16* __restrict__ Qb, const __bf16* __restrict__ Kb,
                 const __bf16* __restrict__ Vt, const float* __restrict__ t,
                 const float* __restrict__ gammap, float* __restrict__ out) {
    // per-wave P transpose buffer; stride 72 halves (144B) keeps chunks 16B aligned
    __shared__ __align__(16) __bf16 pbuf[WAVES][16][72];

    const int lane = threadIdx.x & 31;
    const int w    = threadIdx.x >> 5;
    const int gtile = blockIdx.x * WAVES + w;          // global 16-row tile id
    const int b   = gtile / (N_ / 16);
    const int n0  = (gtile % (N_ / 16)) * 16;
    const int hl  = lane >> 4;                         // which 16-lane half
    const int l16 = lane & 15;

    const __bf16* Qbb = Qb + (size_t)b * N_ * KD_;
    const __bf16* Kbb = Kb + (size_t)b * N_ * KD_;
    const __bf16* Vtb = Vt + (size_t)b * D_ * N_;

    // ---- load Q tile as WMMA A operand (16x32 bf16) ----
    U32B uq;
    {
        const __bf16* qrow = Qbb + (size_t)(n0 + l16) * KD_ + hl * 16;
        uq.q[0] = *(const uint4*)(qrow);
        uq.q[1] = *(const uint4*)(qrow + 8);
    }
    const v16bf aq = uq.v;

    // ones B-matrix for row-sum-via-WMMA (softmax denominator)
    v16bf onesb;
#pragma unroll
    for (int i = 0; i < 16; ++i) onesb[i] = (__bf16)1.0f;

    v8f O[D_ / 16];
    v8f Lacc;                         // softmax denominator accumulator (C layout)
    float mrow[8];
#pragma unroll
    for (int j = 0; j < D_ / 16; ++j)
#pragma unroll
        for (int r = 0; r < 8; ++r) O[j][r] = 0.f;
#pragma unroll
    for (int r = 0; r < 8; ++r) { mrow[r] = -1e30f; Lacc[r] = 0.f; }

    // per-lane base addresses (loop-invariant parts)
    const __bf16* kbase = Kbb + (size_t)l16 * KD_ + hl * 16;   // + m*KD_ per tile
    const __bf16* vbase = Vtb + (size_t)l16 * N_ + hl * 16;    // + d*N_ + m0

    for (int m0 = 0; m0 < N_; m0 += 64) {
        // ---- stage all four K B-tiles (distinct regs -> one load clause)
        U32B uk[4];
#pragma unroll
        for (int c = 0; c < 4; ++c) {
            const uint4* p = (const uint4*)(kbase + (size_t)(m0 + 16 * c) * KD_);
            uk[c].q[0] = p[0];
            uk[c].q[1] = p[1];
        }
        v8f z;
#pragma unroll
        for (int r = 0; r < 8; ++r) z[r] = 0.f;
        v8f S[4];
#pragma unroll
        for (int c = 0; c < 4; ++c)
            S[c] = __builtin_amdgcn_wmma_f32_16x16x32_bf16(false, aq, false, uk[c].v, (short)0, z, false, false);

        // ---- online softmax (exp2 domain; log2e folded into Q)
        v8f scv;
#pragma unroll
        for (int r = 0; r < 8; ++r) {
            float mx = fmaxf(fmaxf(S[0][r], S[1][r]), fmaxf(S[2][r], S[3][r]));
            mx = rowmax16(mx);
            float mnew = fmaxf(mrow[r], mx);
            scv[r] = exp2f(mrow[r] - mnew);
            mrow[r] = mnew;
            float p0 = exp2f(S[0][r] - mnew);
            float p1 = exp2f(S[1][r] - mnew);
            float p2 = exp2f(S[2][r] - mnew);
            float p3 = exp2f(S[3][r] - mnew);
            // P (f32, C layout) -> bf16 A layout via per-wave LDS transpose
            pbuf[w][hl * 8 + r][l16]      = (__bf16)p0;
            pbuf[w][hl * 8 + r][l16 + 16] = (__bf16)p1;
            pbuf[w][hl * 8 + r][l16 + 32] = (__bf16)p2;
            pbuf[w][hl * 8 + r][l16 + 48] = (__bf16)p3;
        }
#pragma unroll
        for (int j = 0; j < D_ / 16; ++j) O[j] *= scv;

        U32B up0, up1;   // same-wave RAW; compiler inserts s_wait_dscnt
        up0.q[0] = *(const uint4*)&pbuf[w][l16][hl * 16];
        up0.q[1] = *(const uint4*)&pbuf[w][l16][8 + hl * 16];
        up1.q[0] = *(const uint4*)&pbuf[w][l16][32 + hl * 16];
        up1.q[1] = *(const uint4*)&pbuf[w][l16][40 + hl * 16];
        const v16bf ap0 = up0.v;
        const v16bf ap1 = up1.v;

        // ---- softmax denominator: row-sum of P chunk via WMMA with ones B
        Lacc = __builtin_amdgcn_wmma_f32_16x16x32_bf16(false, ap0, false, onesb, (short)0, Lacc * scv, false, false);
        Lacc = __builtin_amdgcn_wmma_f32_16x16x32_bf16(false, ap1, false, onesb, (short)0, Lacc, false, false);

        // ---- O += P * V : d-tiles in pairs; 4 staged B-tiles then 4 wmma
#pragma unroll
        for (int j = 0; j < D_ / 16; j += 2) {
            const uint4* pv0 = (const uint4*)(vbase + (size_t)(j * 16) * N_ + m0);
            const uint4* pv1 = (const uint4*)(vbase + (size_t)(j * 16 + 16) * N_ + m0);
            U32B uv[4];
            uv[0].q[0] = pv0[0]; uv[0].q[1] = pv0[1];
            uv[1].q[0] = pv0[4]; uv[1].q[1] = pv0[5];
            uv[2].q[0] = pv1[0]; uv[2].q[1] = pv1[1];
            uv[3].q[0] = pv1[4]; uv[3].q[1] = pv1[5];
            O[j]     = __builtin_amdgcn_wmma_f32_16x16x32_bf16(false, ap0, false, uv[0].v, (short)0, O[j],     false, false);
            O[j]     = __builtin_amdgcn_wmma_f32_16x16x32_bf16(false, ap1, false, uv[1].v, (short)0, O[j],     false, false);
            O[j + 1] = __builtin_amdgcn_wmma_f32_16x16x32_bf16(false, ap0, false, uv[2].v, (short)0, O[j + 1], false, false);
            O[j + 1] = __builtin_amdgcn_wmma_f32_16x16x32_bf16(false, ap1, false, uv[3].v, (short)0, O[j + 1], false, false);
        }
    }

    // ---- epilogue: out = gamma * (O / l) + t
    float gamma = gammap[0];
    v8f invl;
#pragma unroll
    for (int r = 0; r < 8; ++r) invl[r] = 1.f / Lacc[r];
    const float* tb = t   + (size_t)b * N_ * D_;
    float*       ob = out + (size_t)b * N_ * D_;
#pragma unroll
    for (int j = 0; j < D_ / 16; ++j) {
#pragma unroll
        for (int r = 0; r < 8; ++r) {
            int n = n0 + hl * 8 + r;
            int d = j * 16 + l16;
            size_t off = (size_t)n * D_ + d;
            ob[off] = gamma * (O[j][r] * invl[r]) + tb[off];
        }
    }
}

// ---------------------------------------------------------------------------
extern "C" void kernel_launch(void* const* d_in, const int* in_sizes, int n_in,
                              void* d_out, int out_size, void* d_ws, size_t ws_size,
                              hipStream_t stream) {
    const float* x     = (const float*)d_in[0];
    const float* t     = (const float*)d_in[1];
    const float* Wf    = (const float*)d_in[2];
    const float* bf    = (const float*)d_in[3];
    const float* Wg    = (const float*)d_in[4];
    const float* bg    = (const float*)d_in[5];
    const float* gamma = (const float*)d_in[6];
    float* out = (float*)d_out;

    // workspace: Qb (1MB) | Kb (1MB) | Vt (8MB), all bf16
    __bf16* Qb = (__bf16*)d_ws;
    __bf16* Kb = Qb + (size_t)B_ * N_ * KD_;
    __bf16* Vt = Kb + (size_t)B_ * N_ * KD_;

    fg_kernel<<<(B_ * N_) / 256, 256, 0, stream>>>(x, Wf, bf, Wg, bg, Qb, Kb);

    dim3 tb(32, 8, 1), tg(N_ / 32, D_ / 32, B_);
    vt_kernel<<<tg, tb, 0, stream>>>(t, Vt);

    attn_kernel<<<(B_ * (N_ / 16)) / WAVES, 32 * WAVES, 0, stream>>>(Qb, Kb, Vt, t, gamma, out);
}
